// ImBert_29841432772937
// MI455X (gfx1250) — matrix operand order repaired and checked by reading, not compiled, not run on hardware
//
#include <hip/hip_runtime.h>
#include <hip/hip_bf16.h>
#include <math.h>

typedef __attribute__((ext_vector_type(16))) _Float16 v16h;
typedef __attribute__((ext_vector_type(2)))  _Float16 v2h;
typedef __attribute__((ext_vector_type(8)))  float    v8f;
typedef __attribute__((ext_vector_type(4)))  unsigned u32x4;
typedef __attribute__((ext_vector_type(8)))  int      i32x8;
typedef __attribute__((ext_vector_type(4)))  int      i32x4;

#define Bc   64
#define Sc   512
#define Fc   75
#define NMc  153
#define NNEGc 10
#define Lc   4
#define Hc   3
#define NVc  320
#define Gc   3
#define VDc  25
#define DFFc 300

#define FP    96    // F padded to 3 K-steps of 32
#define DHP   32    // per-head dim padded
#define DFFP  320   // DFF padded
#define QKVLD 288   // 3 * FP
#define BSc   (Bc*Sc)
#define BTc   (Bc*NMc)

// ---------------- WMMA fragment helpers (CDNA5 wave32 layouts) ----------------

// A-matrix 16x32 f16: lanes 0-15 = rows, VGPR i / half hh: K = hh*16 + 2i
__device__ inline v16h load_fragA(const _Float16* base, int ld, int mBase, int kBase, int lane) {
  int m = lane & 15, hh = lane >> 4;
  const _Float16* p = base + (size_t)(mBase + m) * ld + kBase;
  v16h f;
#pragma unroll
  for (int i = 0; i < 8; ++i) {
    int k = hh * 16 + 2 * i;
    v2h t = *(const v2h*)(p + k);
    f[2 * i] = t[0]; f[2 * i + 1] = t[1];
  }
  return f;
}

// B-matrix 32x16 f16 (stored transposed [N][K]): lane = column n, K = hh*16 + 2i
__device__ inline v16h load_fragB(const _Float16* base, int ld, int nBase, int kBase, int lane) {
  int n = lane & 15, hh = lane >> 4;
  const _Float16* p = base + (size_t)(nBase + n) * ld + kBase + hh * 16;
  v16h f;
#pragma unroll
  for (int i = 0; i < 8; ++i) {
    v2h t = *(const v2h*)(p + 2 * i);
    f[2 * i] = t[0]; f[2 * i + 1] = t[1];
  }
  return f;
}

__device__ inline v8f wmma32(v16h a, v16h b, v8f c) {
  return __builtin_amdgcn_wmma_f32_16x16x32_f16(false, a, false, b, (short)0, c, false, false);
}

// ---------------- Generic WMMA GEMM: C = A[M,Kp] * Wt[N,Kp]^T + bias ----------------
// Weight panel for this block's 16 output columns is DMA'd to LDS by the
// Tensor Data Mover (TENSOR_LOAD_TO_LDS, TENSORcnt), then all 4 waves read
// B fragments from LDS; A fragments stream from global (L2-resident).
// EPI: 0 = store f16, 1 = exact-GELU then f16, 2 = store f32
template<int EPI>
__global__ __launch_bounds__(128)
void k_gemm(const _Float16* __restrict__ A, const _Float16* __restrict__ Wt,
            const float* __restrict__ bias, void* __restrict__ outp,
            int M, int N, int Kp, int ldOut) {
  __shared__ _Float16 ldsW[16 * DFFP];   // max panel: 16 x 320 f16 = 10 KB, LDS offset 0
  int lane = threadIdx.x & 31;
  int wave = threadIdx.x >> 5;
  int nBase = blockIdx.x * 16;
  int mBase = (blockIdx.y * 4 + wave) * 16;

  if (wave == 0) {
    // Build 2D tensor D# (CDNA5 ISA): tile Kp x 16 of tensor Kp x N.
    unsigned long long ga = (unsigned long long)(const void*)(Wt + (size_t)nBase * Kp);
    u32x4 g0;
    g0[0] = 1u;                                                   // count=1 user descriptor
    g0[1] = 0u;                                                   // lds_addr (ldsW @ 0)
    g0[2] = (unsigned)ga;                                         // global_addr[31:0]
    g0[3] = (unsigned)((ga >> 32) & 0x01ffffffull) | 0x80000000u; // addr[56:32] | type=2
    unsigned td0 = (unsigned)Kp, td1 = (unsigned)N;
    unsigned tile0 = (unsigned)Kp, tile1 = 16u;
    unsigned long long st0 = (unsigned long long)Kp;
    i32x8 g1;
    g1[0] = (int)(1u << 16);                                  // data_size = 1 -> 2 bytes
    g1[1] = (int)((td0 & 0xffffu) << 16);                     // tensor_dim0[15:0]
    g1[2] = (int)((td0 >> 16) | ((td1 & 0xffffu) << 16));     // dim0[31:16] | dim1[15:0]
    g1[3] = (int)((td1 >> 16) | (tile0 << 16));               // dim1[31:16] | tile_dim0
    g1[4] = (int)(tile1 & 0xffffu);                           // tile_dim1 (tile_dim2 = 0)
    g1[5] = (int)(st0 & 0xffffffffull);                       // tensor_dim0_stride[31:0]
    g1[6] = (int)((st0 >> 32) & 0xffffu);                     // stride0[47:32] | stride1 lo
    g1[7] = 0;
    i32x4 gz;
    gz[0] = 0; gz[1] = 0; gz[2] = 0; gz[3] = 0;
    i32x8 gz8;
    gz8[0] = 0; gz8[1] = 0; gz8[2] = 0; gz8[3] = 0;
    gz8[4] = 0; gz8[5] = 0; gz8[6] = 0; gz8[7] = 0;
    __builtin_amdgcn_tensor_load_to_lds(g0, g1, gz, gz, gz8, 0);
    __builtin_amdgcn_s_wait_tensorcnt(0);
  }
  __syncthreads();

  __builtin_prefetch(A + (size_t)mBase * Kp, 0, 0);
  v8f acc = {};
  for (int k = 0; k < Kp; k += 32) {
    v16h a = load_fragA(A, Kp, mBase, k, lane);
    v16h b = load_fragB(ldsW, Kp, 0, k, lane);
    acc = wmma32(a, b, acc);
  }
  if (mBase < M && nBase < N) {
    int n = nBase + (lane & 15);
    int mOff = mBase + ((lane >> 4) << 3);
    float bv = bias[n];
#pragma unroll
    for (int j = 0; j < 8; ++j) {
      float v = acc[j] + bv;
      int row = mOff + j;
      if (EPI == 1) v = 0.5f * v * (1.0f + erff(v * 0.70710678118f));
      if (EPI == 2) ((float*)outp)[(size_t)row * ldOut + n] = v;
      else          ((_Float16*)outp)[(size_t)row * ldOut + n] = (_Float16)v;
    }
  }
}

// ---------------- Weight / bias padding (with optional per-head block mapping) ----------
__global__ void k_pad_wt(const float* __restrict__ src, _Float16* __restrict__ dst,
                         int dstN, int dstK, int Nreal, int Kreal,
                         int nBlkPad, int nBlkReal, int kBlkPad, int kBlkReal, int srcLd) {
  int idx = blockIdx.x * blockDim.x + threadIdx.x;
  if (idx >= dstN * dstK) return;
  int o = idx / dstK, i = idx % dstK;
  int oh = o / nBlkPad, od = o % nBlkPad;
  int ih = i / kBlkPad, id = i % kBlkPad;
  int oreal = oh * nBlkReal + od, ireal = ih * kBlkReal + id;
  float v = 0.f;
  if (od < nBlkReal && id < kBlkReal && oreal < Nreal && ireal < Kreal)
    v = src[(size_t)ireal * srcLd + oreal];
  dst[idx] = (_Float16)v;
}

__global__ void k_pad_bias(const float* __restrict__ src, float* __restrict__ dst,
                           int dstN, int Nreal, int nBlkPad, int nBlkReal) {
  int o = blockIdx.x * blockDim.x + threadIdx.x;
  if (o >= dstN) return;
  int oh = o / nBlkPad, od = o % nBlkPad;
  int oreal = oh * nBlkReal + od;
  dst[o] = (od < nBlkReal && oreal < Nreal) ? src[oreal] : 0.f;
}

// ---------------- Front-end kernels ----------------
__global__ void k_scatter(const int* __restrict__ mask_idx, int* __restrict__ flags) {
  int i = blockIdx.x * blockDim.x + threadIdx.x;
  if (i >= Bc * NMc) return;
  flags[(i / NMc) * Sc + mask_idx[i]] = 1;
}

__global__ void k_ln0(const float* __restrict__ poses, const float* __restrict__ g,
                      const float* __restrict__ bb, const float* __restrict__ mask_embed,
                      const int* __restrict__ flags, float* __restrict__ y0,
                      float* __restrict__ y) {
  int r = blockIdx.x * blockDim.x + threadIdx.x;
  if (r >= BSc) return;
  const float* x = poses + (size_t)r * Fc;
  float mean = 0.f;
  for (int f = 0; f < Fc; ++f) mean += x[f];
  mean *= (1.0f / Fc);
  float var = 0.f;
  for (int f = 0; f < Fc; ++f) { float d = x[f] - mean; var += d * d; }
  var *= (1.0f / Fc);
  float inv = rsqrtf(var + 1e-5f);
  int msk = flags[r];
  for (int f = 0; f < Fc; ++f) {
    float v = (x[f] - mean) * inv * g[f] + bb[f];
    y0[(size_t)r * Fc + f] = v;
    y[(size_t)r * Fc + f] = msk ? mask_embed[f] : v;
  }
}

// relative positional conv, k=9, pad 4; y2 = y + conv(y) + b_pos
__global__ void k_conv(const float* __restrict__ y, const float* __restrict__ w_pos,
                       const float* __restrict__ b_pos, float* __restrict__ y2) {
  int idx = blockIdx.x * blockDim.x + threadIdx.x;
  if (idx >= BSc * Fc) return;
  int o = idx % Fc, s = (idx / Fc) % Sc, b = idx / (Fc * Sc);
  float acc = 0.f;
  for (int k = 0; k < 9; ++k) {
    int sp = s + k - 4;
    if (sp < 0 || sp >= Sc) continue;
    const float* yr = y + ((size_t)b * Sc + sp) * Fc;
    const float* wr = w_pos + ((size_t)o * Fc) * 9 + k;
    for (int i = 0; i < Fc; ++i) acc += yr[i] * wr[(size_t)i * 9];
  }
  y2[idx] = y[idx] + acc + b_pos[o];
}

__global__ void k_embed_ln(const float* __restrict__ y2, const float* __restrict__ pos_table,
                           const float* __restrict__ tok_emb, const float* __restrict__ g,
                           const float* __restrict__ bb, float* __restrict__ h,
                           _Float16* __restrict__ h16) {
  int r = blockIdx.x * blockDim.x + threadIdx.x;
  if (r >= BSc) return;
  int s = r % Sc;
  const float* x = y2 + (size_t)r * Fc;
  const float* pt = pos_table + (size_t)s * Fc;
  float mean = 0.f;
  for (int f = 0; f < Fc; ++f) mean += x[f] + pt[f] + tok_emb[f];
  mean *= (1.0f / Fc);
  float var = 0.f;
  for (int f = 0; f < Fc; ++f) { float d = x[f] + pt[f] + tok_emb[f] - mean; var += d * d; }
  var *= (1.0f / Fc);
  float inv = rsqrtf(var + 1e-12f);
  for (int f = 0; f < Fc; ++f) {
    float v = (x[f] + pt[f] + tok_emb[f] - mean) * inv * g[f] + bb[f];
    h[(size_t)r * Fc + f] = v;
    h16[(size_t)r * FP + f] = (_Float16)v;
  }
  for (int f = Fc; f < FP; ++f) h16[(size_t)r * FP + f] = (_Float16)0.f;
}

// h = LN(h + tmp[:, :75]); also refresh padded f16 copy
__global__ void k_resid_ln(float* __restrict__ h, const float* __restrict__ tmp,
                           const float* __restrict__ g, const float* __restrict__ bb,
                           _Float16* __restrict__ h16) {
  int r = blockIdx.x * blockDim.x + threadIdx.x;
  if (r >= BSc) return;
  const float* t = tmp + (size_t)r * FP;
  float* hr = h + (size_t)r * Fc;
  float mean = 0.f;
  for (int f = 0; f < Fc; ++f) mean += hr[f] + t[f];
  mean *= (1.0f / Fc);
  float var = 0.f;
  for (int f = 0; f < Fc; ++f) { float d = hr[f] + t[f] - mean; var += d * d; }
  var *= (1.0f / Fc);
  float inv = rsqrtf(var + 1e-12f);
  for (int f = 0; f < Fc; ++f) {
    float v = (hr[f] + t[f] - mean) * inv * g[f] + bb[f];
    hr[f] = v;
    h16[(size_t)r * FP + f] = (_Float16)v;
  }
  for (int f = Fc; f < FP; ++f) h16[(size_t)r * FP + f] = (_Float16)0.f;
}

// ---------------- Flash-style attention: one block = (b, head, 16-query tile) ----------
__global__ __launch_bounds__(128)
void k_attn(const _Float16* __restrict__ qkv16, _Float16* __restrict__ attn16) {
  __shared__ float sc[16][Sc];
  __shared__ _Float16 p16[16][Sc];
  __shared__ float red[16][8];
  __shared__ float rowv[16];
  int lane = threadIdx.x & 31, wave = threadIdx.x >> 5;
  int qt = blockIdx.x;                 // 0..31 query tiles
  int b = blockIdx.y / Hc, head = blockIdx.y % Hc;
  size_t rowBase = (size_t)b * Sc * QKVLD;
  const _Float16* qbase = qkv16 + rowBase + head * DHP;           // Q at col offset 0
  const _Float16* kbase = qkv16 + rowBase + FP + head * DHP;      // K at col offset 96

  // scores: S[16,512] = Q(16x32) * K^T, scaled by 1/sqrt(25)
  v16h qa = load_fragA(qbase, QKVLD, qt * 16, 0, lane);
  int n = lane & 15, hh = lane >> 4;
  for (int nt = wave; nt < Sc / 16; nt += 4) {
    v16h kb = load_fragB(kbase, QKVLD, nt * 16, 0, lane);
    v8f acc = {};
    acc = wmma32(qa, kb, acc);
#pragma unroll
    for (int j = 0; j < 8; ++j) sc[hh * 8 + j][nt * 16 + n] = acc[j] * 0.2f;
  }
  __syncthreads();

  // softmax over 512 keys, 128 threads = 16 rows x 8 segments of 64
  int r = threadIdx.x & 15, seg = threadIdx.x >> 4;
  float m = -1e30f;
  for (int c = seg * 64; c < seg * 64 + 64; ++c) m = fmaxf(m, sc[r][c]);
  red[r][seg] = m; __syncthreads();
  if (seg == 0) {
    float mm = red[r][0];
    for (int k = 1; k < 8; ++k) mm = fmaxf(mm, red[r][k]);
    rowv[r] = mm;
  }
  __syncthreads();
  float mm = rowv[r], ssum = 0.f;
  for (int c = seg * 64; c < seg * 64 + 64; ++c) {
    float e = expf(sc[r][c] - mm); sc[r][c] = e; ssum += e;
  }
  red[r][seg] = ssum; __syncthreads();
  if (seg == 0) {
    float t = 0.f;
    for (int k = 0; k < 8; ++k) t += red[r][k];
    rowv[r] = 1.f / t;
  }
  __syncthreads();
  float invs = rowv[r];
  for (int c = seg * 64; c < seg * 64 + 64; ++c) p16[r][c] = (_Float16)(sc[r][c] * invs);
  __syncthreads();

  // O(16x32) = P(16x512) * V(512x32); waves 0,1 each take 16 dh columns
  if (wave < 2) {
    v8f acc = {};
    const _Float16* vb = qkv16 + rowBase + 2 * FP + head * DHP + wave * 16 + n;
    for (int kk = 0; kk < Sc / 32; ++kk) {
      v16h pa = load_fragA(&p16[0][0], Sc, 0, kk * 32, lane);
      v16h vf;
#pragma unroll
      for (int i = 0; i < 8; ++i) {
        int k0 = kk * 32 + hh * 16 + 2 * i;
        vf[2 * i]     = vb[(size_t)k0 * QKVLD];
        vf[2 * i + 1] = vb[(size_t)(k0 + 1) * QKVLD];
      }
      acc = wmma32(pa, vf, acc);
    }
#pragma unroll
    for (int j = 0; j < 8; ++j)
      attn16[((size_t)b * Sc + qt * 16 + hh * 8 + j) * FP + head * DHP + wave * 16 + n] =
          (_Float16)acc[j];
  }
}

// ---------------- Back-end kernels ----------------
__global__ void k_gather(const float* __restrict__ h, const float* __restrict__ y0,
                         const int* __restrict__ mask_idx, float* __restrict__ fe_out,
                         _Float16* __restrict__ ym16) {
  int idx = blockIdx.x * blockDim.x + threadIdx.x;
  if (idx >= BTc * Fc) return;
  int f = idx % Fc, rt = idx / Fc;
  int b = rt / NMc, t = rt % NMc;
  int s = mask_idx[b * NMc + t];
  float v = h[((size_t)b * Sc + s) * Fc + f];
  fe_out[idx] = v;
  ym16[(size_t)rt * FP + f] = (_Float16)y0[((size_t)b * Sc + s) * Fc + f];
}

__global__ void k_vq(const float* __restrict__ logits, const float* __restrict__ gumbel,
                     const float* __restrict__ codebook, float* __restrict__ avgp,
                     _Float16* __restrict__ quant16) {
  int idx = blockIdx.x * blockDim.x + threadIdx.x;
  if (idx >= BTc * Gc) return;
  int g = idx % Gc, r = idx / Gc;
  const float* lr = logits + (size_t)r * (Gc * NVc) + g * NVc;
  const float* gr = gumbel + (size_t)r * (Gc * NVc) + g * NVc;
  float m = -1e30f, mg = -1e30f; int arg = 0;
  for (int nn = 0; nn < NVc; ++nn) {
    float l = lr[nn];
    m = fmaxf(m, l);
    float gl = (l + gr[nn]) * 0.5f;           // TAU = 2
    if (gl > mg) { mg = gl; arg = nn; }
  }
  float s = 0.f;
  for (int nn = 0; nn < NVc; ++nn) s += expf(lr[nn] - m);
  float inv = 1.f / (s * (float)BTc);
  for (int nn = 0; nn < NVc; ++nn)
    atomicAdd(&avgp[g * NVc + nn], expf(lr[nn] - m) * inv);
  for (int d = 0; d < VDc; ++d)
    quant16[(size_t)r * FP + g * VDc + d] = (_Float16)codebook[(size_t)arg * VDc + d];
}

__global__ void k_contrast(const float* __restrict__ fe, const float* __restrict__ qrep,
                           const int* __restrict__ neg_idxs, float* __restrict__ accum) {
  int r = blockIdx.x * blockDim.x + threadIdx.x;
  if (r >= BTc) return;
  int b = r / NMc, t = r % NMc;
  float fv[Fc], pv[Fc];
  float na = 0.f;
  for (int f = 0; f < Fc; ++f) { fv[f] = fe[(size_t)r * Fc + f]; na += fv[f] * fv[f]; }
  na = fmaxf(sqrtf(na), 1e-8f);
  float nb = 0.f, dot = 0.f;
  for (int f = 0; f < Fc; ++f) {
    pv[f] = qrep[(size_t)r * FP + f];
    nb += pv[f] * pv[f]; dot += fv[f] * pv[f];
  }
  float l0 = dot / (na * fmaxf(sqrtf(nb), 1e-8f)) * 10.f;
  float mx = l0;
  float lj[NNEGc];
  for (int j = 0; j < NNEGc; ++j) {
    int nt = neg_idxs[(size_t)b * (NMc * NNEGc) + t * NNEGc + j];
    const float* qn = qrep + (size_t)(b * NMc + nt) * FP;
    float d2 = 0.f, n2 = 0.f; bool eq = true;
    for (int f = 0; f < Fc; ++f) {
      float q = qn[f];
      d2 += fv[f] * q; n2 += q * q;
      eq = eq && (q == pv[f]);
    }
    float lv = eq ? -1e30f : d2 / (na * fmaxf(sqrtf(n2), 1e-8f)) * 10.f;
    lj[j] = lv; mx = fmaxf(mx, lv);
  }
  float se = expf(l0 - mx);
  for (int j = 0; j < NNEGc; ++j) se += expf(lj[j] - mx);
  atomicAdd(accum, -(l0 - mx - logf(se)));
}

__global__ void k_final(const float* __restrict__ accum, const float* __restrict__ avgp,
                        float* __restrict__ out2) {
  float perp = 0.f;
  for (int g = 0; g < Gc; ++g) {
    float hsum = 0.f;
    for (int nn = 0; nn < NVc; ++nn) {
      float p = avgp[g * NVc + nn];
      hsum += p * logf(p + 1e-7f);
    }
    perp += expf(-hsum);
  }
  out2[0] = accum[0] / (float)BTc;
  out2[1] = 1.f - perp;
}

// ---------------- Host orchestration ----------------
extern "C" void kernel_launch(void* const* d_in, const int* in_sizes, int n_in,
                              void* d_out, int out_size, void* d_ws, size_t ws_size,
                              hipStream_t stream) {
  (void)in_sizes; (void)n_in; (void)out_size; (void)ws_size;
  const float* poses     = (const float*)d_in[0];
  const int*   mask_idx  = (const int*)d_in[1];
  const int*   neg_idxs  = (const int*)d_in[2];
  const float* gumbel    = (const float*)d_in[3];
  const float* ln_g      = (const float*)d_in[4];
  const float* ln_b      = (const float*)d_in[5];
  const float* mask_emb  = (const float*)d_in[6];
  const float* w_pos     = (const float*)d_in[7];
  const float* b_pos     = (const float*)d_in[8];
  const float* pos_table = (const float*)d_in[9];
  const float* tok_emb   = (const float*)d_in[10];
  const float* emb_ln_g  = (const float*)d_in[11];
  const float* emb_ln_b  = (const float*)d_in[12];
  const float* qkv_w     = (const float*)d_in[13];
  const float* qkv_b     = (const float*)d_in[14];
  const float* attn_o_w  = (const float*)d_in[15];
  const float* attn_o_b  = (const float*)d_in[16];
  const float* ln1_g     = (const float*)d_in[17];
  const float* ln1_b     = (const float*)d_in[18];
  const float* ffn_w1    = (const float*)d_in[19];
  const float* ffn_b1    = (const float*)d_in[20];
  const float* ffn_w2    = (const float*)d_in[21];
  const float* ffn_b2    = (const float*)d_in[22];
  const float* ln2_g     = (const float*)d_in[23];
  const float* ln2_b     = (const float*)d_in[24];
  const float* vq_w      = (const float*)d_in[25];
  const float* vq_b      = (const float*)d_in[26];
  const float* codebook  = (const float*)d_in[27];
  const float* proj_w    = (const float*)d_in[28];
  const float* proj_b    = (const float*)d_in[29];

  char* base = (char*)d_ws;
  size_t off = 0;
  auto carve = [&](size_t bytes) -> void* {
    void* p = base + off;
    off = (off + bytes + 255) & ~(size_t)255;
    return p;
  };

  int*       flags  = (int*)carve((size_t)BSc * 4);
  float*     y0     = (float*)carve((size_t)BSc * Fc * 4);
  float*     ybuf   = (float*)carve((size_t)BSc * Fc * 4);
  float*     y2     = (float*)carve((size_t)BSc * Fc * 4);
  float*     h      = (float*)carve((size_t)BSc * Fc * 4);
  _Float16*  h16    = (_Float16*)carve((size_t)BSc * FP * 2);
  _Float16*  qkv16  = (_Float16*)carve((size_t)BSc * QKVLD * 2);
  _Float16*  attn16 = (_Float16*)carve((size_t)BSc * FP * 2);
  float*     tmp32  = (float*)carve((size_t)BSc * FP * 4);
  _Float16*  ffn16  = (_Float16*)carve((size_t)BSc * DFFP * 2);
  _Float16*  wqkv   = (_Float16*)carve((size_t)Lc * QKVLD * FP * 2);
  float*     bqkv   = (float*)carve((size_t)Lc * QKVLD * 4);
  _Float16*  wo     = (_Float16*)carve((size_t)Lc * FP * FP * 2);
  float*     bo     = (float*)carve((size_t)Lc * FP * 4);
  _Float16*  w1     = (_Float16*)carve((size_t)Lc * DFFP * FP * 2);
  float*     pb1    = (float*)carve((size_t)Lc * DFFP * 4);
  _Float16*  w2     = (_Float16*)carve((size_t)Lc * FP * DFFP * 2);
  float*     pb2    = (float*)carve((size_t)Lc * FP * 4);
  _Float16*  wvq    = (_Float16*)carve((size_t)(Gc * NVc) * FP * 2);
  float*     bvq    = (float*)carve((size_t)(Gc * NVc) * 4);
  _Float16*  wproj  = (_Float16*)carve((size_t)FP * FP * 2);
  float*     bproj  = (float*)carve((size_t)FP * 4);
  _Float16*  ym16   = (_Float16*)carve((size_t)BTc * FP * 2);
  float*     logits = (float*)carve((size_t)BTc * Gc * NVc * 4);
  float*     avgp   = (float*)carve((size_t)Gc * NVc * 4);
  _Float16*  q16    = (_Float16*)carve((size_t)BTc * FP * 2);
  float*     qrep   = (float*)carve((size_t)BTc * FP * 4);
  float*     accum  = (float*)carve(256);

  // zero-init buffers whose padding / accumulation must start clean
  (void)hipMemsetAsync(flags, 0, (size_t)BSc * 4, stream);
  (void)hipMemsetAsync(avgp, 0, (size_t)Gc * NVc * 4, stream);
  (void)hipMemsetAsync(accum, 0, 4, stream);
  (void)hipMemsetAsync(ym16, 0, (size_t)BTc * FP * 2, stream);
  (void)hipMemsetAsync(q16, 0, (size_t)BTc * FP * 2, stream);

  // ---- weight prep (padded, transposed to [N][K] f16) ----
  for (int l = 0; l < Lc; ++l) {
    for (int q = 0; q < 3; ++q) {
      k_pad_wt<<<(FP * FP + 255) / 256, 256, 0, stream>>>(
          qkv_w + (size_t)(l * 3 + q) * Fc * Fc, wqkv + (size_t)(l * 3 + q) * FP * FP,
          FP, FP, Fc, Fc, DHP, VDc, FP, Fc, Fc);
      k_pad_bias<<<1, 256, 0, stream>>>(qkv_b + (size_t)(l * 3 + q) * Fc,
                                        bqkv + (size_t)l * QKVLD + q * FP, FP, Fc, DHP, VDc);
    }
    k_pad_wt<<<(FP * FP + 255) / 256, 256, 0, stream>>>(
        attn_o_w + (size_t)l * Fc * Fc, wo + (size_t)l * FP * FP,
        FP, FP, Fc, Fc, FP, Fc, DHP, VDc, Fc);
    k_pad_bias<<<1, 256, 0, stream>>>(attn_o_b + (size_t)l * Fc, bo + (size_t)l * FP,
                                      FP, Fc, FP, Fc);
    k_pad_wt<<<(DFFP * FP + 255) / 256, 256, 0, stream>>>(
        ffn_w1 + (size_t)l * Fc * DFFc, w1 + (size_t)l * DFFP * FP,
        DFFP, FP, DFFc, Fc, DFFP, DFFc, FP, Fc, DFFc);
    k_pad_bias<<<2, 256, 0, stream>>>(ffn_b1 + (size_t)l * DFFc, pb1 + (size_t)l * DFFP,
                                      DFFP, DFFc, DFFP, DFFc);
    k_pad_wt<<<(FP * DFFP + 255) / 256, 256, 0, stream>>>(
        ffn_w2 + (size_t)l * DFFc * Fc, w2 + (size_t)l * FP * DFFP,
        FP, DFFP, Fc, DFFc, FP, Fc, DFFP, DFFc, Fc);
    k_pad_bias<<<1, 256, 0, stream>>>(ffn_b2 + (size_t)l * Fc, pb2 + (size_t)l * FP,
                                      FP, Fc, FP, Fc);
  }
  k_pad_wt<<<(Gc * NVc * FP + 255) / 256, 256, 0, stream>>>(
      vq_w, wvq, Gc * NVc, FP, Gc * NVc, Fc, Gc * NVc, Gc * NVc, FP, Fc, Gc * NVc);
  k_pad_bias<<<4, 256, 0, stream>>>(vq_b, bvq, Gc * NVc, Gc * NVc, Gc * NVc, Gc * NVc);
  k_pad_wt<<<(FP * FP + 255) / 256, 256, 0, stream>>>(
      proj_w, wproj, FP, FP, Fc, Fc, FP, Fc, FP, Fc, Fc);
  k_pad_bias<<<1, 256, 0, stream>>>(proj_b, bproj, FP, Fc, FP, Fc);

  // ---- front end ----
  k_scatter<<<(Bc * NMc + 255) / 256, 256, 0, stream>>>(mask_idx, flags);
  k_ln0<<<(BSc + 255) / 256, 256, 0, stream>>>(poses, ln_g, ln_b, mask_emb, flags, y0, ybuf);
  k_conv<<<((size_t)BSc * Fc + 255) / 256, 256, 0, stream>>>(ybuf, w_pos, b_pos, y2);
  k_embed_ln<<<(BSc + 255) / 256, 256, 0, stream>>>(y2, pos_table, tok_emb,
                                                    emb_ln_g, emb_ln_b, h, h16);

  // ---- transformer layers ----
  for (int l = 0; l < Lc; ++l) {
    k_gemm<0><<<dim3(QKVLD / 16, BSc / 64), 128, 0, stream>>>(
        h16, wqkv + (size_t)l * 3 * FP * FP, bqkv + (size_t)l * QKVLD,
        qkv16, BSc, QKVLD, FP, QKVLD);
    k_attn<<<dim3(Sc / 16, Bc * Hc), 128, 0, stream>>>(qkv16, attn16);
    k_gemm<2><<<dim3(FP / 16, BSc / 64), 128, 0, stream>>>(
        attn16, wo + (size_t)l * FP * FP, bo + (size_t)l * FP,
        tmp32, BSc, FP, FP, FP);
    k_resid_ln<<<(BSc + 255) / 256, 256, 0, stream>>>(
        h, tmp32, ln1_g + (size_t)l * Fc, ln1_b + (size_t)l * Fc, h16);
    k_gemm<1><<<dim3(DFFP / 16, BSc / 64), 128, 0, stream>>>(
        h16, w1 + (size_t)l * DFFP * FP, pb1 + (size_t)l * DFFP,
        ffn16, BSc, DFFP, FP, DFFP);
    k_gemm<2><<<dim3(FP / 16, BSc / 64), 128, 0, stream>>>(
        ffn16, w2 + (size_t)l * FP * DFFP, pb2 + (size_t)l * FP,
        tmp32, BSc, FP, DFFP, FP);
    k_resid_ln<<<(BSc + 255) / 256, 256, 0, stream>>>(
        h, tmp32, ln2_g + (size_t)l * Fc, ln2_b + (size_t)l * Fc, h16);
  }

  // ---- back end ----
  float* fe_out = (float*)d_out;                       // [B, NM, F]
  k_gather<<<((size_t)BTc * Fc + 255) / 256, 256, 0, stream>>>(h, y0, mask_idx, fe_out, ym16);
  k_gemm<2><<<dim3(Gc * NVc / 16, BTc / 64), 128, 0, stream>>>(
      ym16, wvq, bvq, logits, BTc, Gc * NVc, FP, Gc * NVc);
  k_vq<<<(BTc * Gc + 255) / 256, 256, 0, stream>>>(logits, gumbel, codebook, avgp, q16);
  k_gemm<2><<<dim3(FP / 16, BTc / 64), 128, 0, stream>>>(
      q16, wproj, bproj, qrep, BTc, FP, FP, FP);
  k_contrast<<<(BTc + 255) / 256, 256, 0, stream>>>(fe_out, qrep, neg_idxs, accum);
  k_final<<<1, 1, 0, stream>>>(accum, avgp, fe_out + (size_t)BTc * Fc);
}
